// modified_77154792506037
// MI455X (gfx1250) — matrix operand order repaired and checked by reading, not compiled, not run on hardware
//
#include <hip/hip_runtime.h>
#include <hip/hip_bf16.h>

typedef __attribute__((ext_vector_type(16))) __bf16 v16bf;
typedef __attribute__((ext_vector_type(8)))  __bf16 v8bf;
typedef __attribute__((ext_vector_type(8)))  float  v8f;
typedef __attribute__((ext_vector_type(4)))  float  v4f;

// Problem constants
constexpr int Bn = 128;          // batch
constexpr int Sn = 196;          // seq
constexpr int En = 768;          // emb
constexpr int Mn = 768;          // out features
constexpr int K_FK  = Sn * En;   // 150528
constexpr int K_MLP = 2 * En;    // 1536
constexpr int KC_FK = 3072;      // per-block K chunk for fk gemm (49 chunks)
constexpr int KSTAGE = 256;      // K staged into LDS per refill
constexpr int LDSPAD = 8;        // bf16 pad to stagger LDS banks
constexpr int NT_COLS = 32;      // block N tile (2 x 16-wide WMMA subtiles)

// ---------------------------------------------------------------------------
// Kernel 0: bulk fp32 -> bf16 conversion (sent_emb copy kept L2-resident)
// ---------------------------------------------------------------------------
__global__ __launch_bounds__(256) void f32_to_bf16_kernel(
    const float* __restrict__ src, __bf16* __restrict__ dst, int n8)
{
    const int i = blockIdx.x * blockDim.x + threadIdx.x;
    if (i >= n8) return;
    const v4f* s = (const v4f*)(src + (size_t)i * 8);
    v4f x0 = __builtin_nontemporal_load(s);       // NT: don't evict bf16 copy
    v4f x1 = __builtin_nontemporal_load(s + 1);
    v8bf o;
#pragma unroll
    for (int j = 0; j < 4; ++j) { o[j] = (__bf16)x0[j]; o[4 + j] = (__bf16)x1[j]; }
    *(v8bf*)(dst + (size_t)i * 8) = o;            // RT store: L2-resident
}

// ---------------------------------------------------------------------------
// Kernel 1: mean-pool the two variable-length spans -> event_pair [B, 2E] bf16
// ---------------------------------------------------------------------------
__global__ __launch_bounds__(256) void extract_events_kernel(
    const float* __restrict__ sent,   // [B, S, E] fp32
    const int* __restrict__ be,       // [B, 4] (s1,e1,s2,e2)
    __bf16* __restrict__ ep)          // [B, 2E] bf16
{
    const int b  = blockIdx.x;
    const int s1 = be[b * 4 + 0], e1 = be[b * 4 + 1];
    const int s2 = be[b * 4 + 2], e2 = be[b * 4 + 3];
    const float inv1 = 1.0f / (float)(e1 - s1);
    const float inv2 = 1.0f / (float)(e2 - s2);
    const float* sb = sent + (size_t)b * Sn * En;
    for (int e = threadIdx.x; e < En; e += blockDim.x) {
        float a1 = 0.0f, a2 = 0.0f;
        for (int s = s1; s < e1; ++s) a1 += sb[(size_t)s * En + e];
        for (int s = s2; s < e2; ++s) a2 += sb[(size_t)s * En + e];
        ep[(size_t)b * (2 * En) + e]      = (__bf16)(a1 * inv1);
        ep[(size_t)b * (2 * En) + En + e] = (__bf16)(a2 * inv2);
    }
}

// ---------------------------------------------------------------------------
// WMMA GEMM: C[128, 32-col tile] (+)= alpha * A[128,K](bf16) @ W[Mtile,K]^T(f32)
// 256 threads = 8 waves; wave w owns output rows 16w..16w+15, both N-subtiles.
// W tile (32 rows x KSTAGE) double-buffered in LDS as bf16 (converted once,
// reused by all 8 waves). Next tile is prefetched to registers (NT) while
// WMMAs consume the current buffer.
// ATOMIC=true : out += alpha*acc (split-K partial)
// ATOMIC=false: out  = alpha*acc + 0.9*bias1[col] + 0.1*bias2[col]
// ---------------------------------------------------------------------------
template <bool ATOMIC>
__global__ __launch_bounds__(256) void wmma_gemm_kernel(
    const __bf16* __restrict__ A, int lda,    // [128, lda] bf16
    const float*  __restrict__ W, int ldw,    // [M, ldw] fp32
    float* __restrict__ out,                  // [128, Mn]
    const float* __restrict__ bias1,
    const float* __restrict__ bias2,
    int kChunk, float alpha)
{
    __shared__ __bf16 ldsB[2][NT_COLS][KSTAGE + LDSPAD];

    const int m0      = blockIdx.x * NT_COLS;
    const int k_begin = blockIdx.y * kChunk;
    const int wave    = threadIdx.x >> 5;     // 0..7
    const int lane    = threadIdx.x & 31;
    const int row0    = wave * 16;            // output rows of this wave
    const int arow    = row0 + (lane & 15);
    const int aoff    = (lane >> 4) << 3;     // +8 K for upper half-wave (A layout)
    const int boff    = (lane >> 4) << 3;     // +8 K for upper half-wave (B layout)

    // staging assignment: 8 threads per W row, 32 K values each
    const int sr = threadIdx.x >> 3;          // 0..31 (row of W tile)
    const int sc = (threadIdx.x & 7) * 32;    // K offset within stage
    const float* wrow = W + (size_t)(m0 + sr) * ldw + (k_begin + sc);

    v8f acc0 = {}, acc1 = {};
    v4f w[8];

    auto fetchW = [&](int ks) {
#pragma unroll
        for (int j = 0; j < 8; ++j)
            w[j] = __builtin_nontemporal_load((const v4f*)(wrow + ks + 4 * j));
    };
    auto stageW = [&](int p) {
#pragma unroll
        for (int h = 0; h < 4; ++h) {
            v8bf pk;
#pragma unroll
            for (int j = 0; j < 4; ++j) {
                pk[j]     = (__bf16)w[2 * h][j];
                pk[4 + j] = (__bf16)w[2 * h + 1][j];
            }
            *(v8bf*)&ldsB[p][sr][sc + 8 * h] = pk;
        }
    };

    fetchW(0);
    stageW(0);
    __syncthreads();

    int p = 0;
    for (int ks = 0; ks < kChunk; ks += KSTAGE) {
        const bool hasNext = (ks + KSTAGE) < kChunk;
        if (hasNext) fetchW(ks + KSTAGE);     // global loads in flight over compute

        const __bf16* abase = A + (size_t)arow * lda + (k_begin + ks + aoff);
#pragma unroll
        for (int kk = 0; kk < KSTAGE; kk += 32) {
            // A operand: 16 bf16, K = {k..k+7, k+16..k+23} (+8 upper half-wave)
            v8bf a0 = *(const v8bf*)(abase + kk);
            v8bf a1 = *(const v8bf*)(abase + kk + 16);
            v16bf av;
#pragma unroll
            for (int j = 0; j < 8; ++j) { av[j] = a0[j]; av[8 + j] = a1[j]; }

            // B operands: 16 bf16, K = {k..k+7, k+16..k+23} (+8 upper half-wave), two N-subtiles
            const __bf16* bp0 = &ldsB[p][lane & 15][kk + boff];
            const __bf16* bp1 = &ldsB[p][16 + (lane & 15)][kk + boff];
            v8bf b00 = *(const v8bf*)(bp0);
            v8bf b01 = *(const v8bf*)(bp0 + 16);
            v8bf b10 = *(const v8bf*)(bp1);
            v8bf b11 = *(const v8bf*)(bp1 + 16);
            v16bf bv0, bv1;
#pragma unroll
            for (int j = 0; j < 8; ++j) {
                bv0[j] = b00[j]; bv0[8 + j] = b01[j];
                bv1[j] = b10[j]; bv1[8 + j] = b11[j];
            }
            acc0 = __builtin_amdgcn_wmma_f32_16x16x32_bf16(
                false, av, false, bv0, (short)0, acc0, false, false);
            acc1 = __builtin_amdgcn_wmma_f32_16x16x32_bf16(
                false, av, false, bv1, (short)0, acc1, false, false);
        }
        if (hasNext) stageW(p ^ 1);
        __syncthreads();
        p ^= 1;
    }

    // ---- store D: row = row0 + r + 8*(lane>=16), col = m0 + (lane&15) ----
    const int col   = m0 + (lane & 15);
    const int rbase = row0 + ((lane >> 4) << 3);
    if (ATOMIC) {
#pragma unroll
        for (int r = 0; r < 8; ++r) {
            atomicAdd(&out[(size_t)(rbase + r) * Mn + col],      alpha * acc0[r]);
            atomicAdd(&out[(size_t)(rbase + r) * Mn + col + 16], alpha * acc1[r]);
        }
    } else {
        const float b0 = 0.9f * bias1[col] + 0.1f * bias2[col];
        const float b1 = 0.9f * bias1[col + 16] + 0.1f * bias2[col + 16];
#pragma unroll
        for (int r = 0; r < 8; ++r) {
            out[(size_t)(rbase + r) * Mn + col]      = alpha * acc0[r] + b0;
            out[(size_t)(rbase + r) * Mn + col + 16] = alpha * acc1[r] + b1;
        }
    }
}

// ---------------------------------------------------------------------------
extern "C" void kernel_launch(void* const* d_in, const int* in_sizes, int n_in,
                              void* d_out, int out_size, void* d_ws, size_t ws_size,
                              hipStream_t stream) {
    const float* sent  = (const float*)d_in[0];   // [128,196,768]
    const int*   be    = (const int*)  d_in[1];   // [128,4]
    const float* mlp_w = (const float*)d_in[2];   // [768,1536]
    const float* mlp_b = (const float*)d_in[3];   // [768]
    const float* fk_w  = (const float*)d_in[4];   // [768,150528]
    const float* fk_b  = (const float*)d_in[5];   // [768]
    float* out = (float*)d_out;                   // [128,768]

    // workspace layout: ep bf16 [128,1536] | sent bf16 [128,150528]
    __bf16* ep      = (__bf16*)d_ws;
    __bf16* sent_bf = (__bf16*)((char*)d_ws + (size_t)Bn * K_MLP * 2);

    // 0) sent_emb fp32 -> bf16 (reused 24x from L2 by the fk GEMM)
    {
        const int n8 = Bn * K_FK / 8;             // 2,408,448 v8 stores
        f32_to_bf16_kernel<<<n8 / 256, 256, 0, stream>>>(sent, sent_bf, n8);
    }

    // 1) span mean-pool -> event_pair (bf16)
    extract_events_kernel<<<Bn, 256, 0, stream>>>(sent, be, ep);

    // 2) pred = 0.9*(event_pair @ mlp_w^T) + 0.9*mlp_b + 0.1*fk_b (initializes d_out)
    wmma_gemm_kernel<false><<<dim3(Mn / NT_COLS, 1), 256, 0, stream>>>(
        ep, K_MLP, mlp_w, K_MLP, out, mlp_b, fk_b, K_MLP, 0.9f);

    // 3) pred += 0.1*(sent_bf16 @ fk_w^T)  (split-K, 24 x 49 blocks, atomic accumulate)
    wmma_gemm_kernel<true><<<dim3(Mn / NT_COLS, K_FK / KC_FK), 256, 0, stream>>>(
        sent_bf, K_FK, fk_w, K_FK, out, nullptr, nullptr, KC_FK, 0.1f);
}
